// NRCA_20512763805918
// MI455X (gfx1250) — matrix operand order repaired and checked by reading, not compiled, not run on hardware
//
#include <hip/hip_runtime.h>
#include <math.h>

// ---------------------------------------------------------------------------
// Types for CDNA5 WMMA (wave32): V_WMMA_F32_16X16X32_F16
// ---------------------------------------------------------------------------
typedef __attribute__((ext_vector_type(16))) _Float16 v16h;
typedef __attribute__((ext_vector_type(8)))  float    v8f;

#define CEILDIV(a, b) (((a) + (b) - 1) / (b))

// Problem constants
#define DM 32
#define H0 96
#define W0 192
#define MAXD 128

// ---------------------------------------------------------------------------
// WMMA GEMM:  Y[N,M] = X[N,K] @ Wt[M,K]^T + bias[M]
// N multiple of 16, M multiple of 32, K multiple of 32 (all guaranteed here).
// One wave computes a 16x32 output tile (two 16x16 WMMA accumulators sharing
// one A fragment -> 2 v_wmma per K-step, half the A traffic).
//
// A-fragment (16-bit A 16x32): lane = hi*16 + m (m = row);
//   elements 0..15 -> K = k0 + 16*hi + e   (row of X contiguous in K)
// B-fragment (16-bit B 32x16): lane = hi*16 + n (n = col);
//   elements 0..15 -> K = k0 + 16*hi + e   (row of Wt contiguous in K)
// D (f32 16x16): lane -> col = lane&15 ; VGPR r -> row = 8*hi + r.
// ---------------------------------------------------------------------------
static __device__ __forceinline__ v16h cvt16(const float* __restrict__ p0,
                                             const float* __restrict__ p1) {
  const float4 x0 = *reinterpret_cast<const float4*>(p0);
  const float4 x1 = *reinterpret_cast<const float4*>(p0 + 4);
  const float4 x2 = *reinterpret_cast<const float4*>(p1);
  const float4 x3 = *reinterpret_cast<const float4*>(p1 + 4);
  v16h r;
  r[0]  = (_Float16)x0.x; r[1]  = (_Float16)x0.y; r[2]  = (_Float16)x0.z; r[3]  = (_Float16)x0.w;
  r[4]  = (_Float16)x1.x; r[5]  = (_Float16)x1.y; r[6]  = (_Float16)x1.z; r[7]  = (_Float16)x1.w;
  r[8]  = (_Float16)x2.x; r[9]  = (_Float16)x2.y; r[10] = (_Float16)x2.z; r[11] = (_Float16)x2.w;
  r[12] = (_Float16)x3.x; r[13] = (_Float16)x3.y; r[14] = (_Float16)x3.z; r[15] = (_Float16)x3.w;
  return r;
}

__global__ __launch_bounds__(256)
void wmma_gemm_kernel(const float* __restrict__ X, const float* __restrict__ Wt,
                      const float* __restrict__ bias, float* __restrict__ Y,
                      int N, int K, int M, int out_trans) {
  const int wave = (blockIdx.x * blockDim.x + threadIdx.x) >> 5;
  const int mt2  = M >> 5;           // column-pair tiles
  const int nt   = N >> 4;
  if (wave >= nt * mt2) return;
  const int rowTile = wave / mt2;
  const int colBase = (wave % mt2) << 5;   // first of 32 output columns

  const int lane = threadIdx.x & 31;
  const int m    = lane & 15;
  const int hi   = lane >> 4;

  const float* __restrict__ arow  = X  + (size_t)(rowTile * 16 + m) * K;
  const float* __restrict__ brow0 = Wt + (size_t)(colBase + m) * K;
  const float* __restrict__ brow1 = Wt + (size_t)(colBase + 16 + m) * K;

  v8f acc0 = {0.f, 0.f, 0.f, 0.f, 0.f, 0.f, 0.f, 0.f};
  v8f acc1 = {0.f, 0.f, 0.f, 0.f, 0.f, 0.f, 0.f, 0.f};

  for (int k0 = 0; k0 < K; k0 += 32) {
    const v16h a  = cvt16(arow + k0 + 16 * hi, arow + k0 + 16 * hi + 8);
    const v16h b0 = cvt16(brow0 + k0 + 16 * hi, brow0 + k0 + 16 * hi + 8);
    const v16h b1 = cvt16(brow1 + k0 + 16 * hi, brow1 + k0 + 16 * hi + 8);
    acc0 = __builtin_amdgcn_wmma_f32_16x16x32_f16(false, a, false, b0,
                                                  (short)0, acc0, false, false);
    acc1 = __builtin_amdgcn_wmma_f32_16x16x32_f16(false, a, false, b1,
                                                  (short)0, acc1, false, false);
  }

  const int col0 = colBase + m;
  const int col1 = col0 + 16;
  const float bv0 = bias[col0];
  const float bv1 = bias[col1];
  const int row0 = rowTile * 16 + hi * 8;

  if (out_trans) {
    // 8 consecutive rows per lane are contiguous -> two b128 stores per tile
    float4 s0, s1;
    s0.x = acc0[0] + bv0; s0.y = acc0[1] + bv0; s0.z = acc0[2] + bv0; s0.w = acc0[3] + bv0;
    s1.x = acc0[4] + bv0; s1.y = acc0[5] + bv0; s1.z = acc0[6] + bv0; s1.w = acc0[7] + bv0;
    float* __restrict__ d0 = Y + (size_t)col0 * N + row0;
    *reinterpret_cast<float4*>(d0)     = s0;
    *reinterpret_cast<float4*>(d0 + 4) = s1;
    s0.x = acc1[0] + bv1; s0.y = acc1[1] + bv1; s0.z = acc1[2] + bv1; s0.w = acc1[3] + bv1;
    s1.x = acc1[4] + bv1; s1.y = acc1[5] + bv1; s1.z = acc1[6] + bv1; s1.w = acc1[7] + bv1;
    float* __restrict__ d1 = Y + (size_t)col1 * N + row0;
    *reinterpret_cast<float4*>(d1)     = s0;
    *reinterpret_cast<float4*>(d1 + 4) = s1;
  } else {
    float* __restrict__ d = Y + (size_t)row0 * M;
#pragma unroll
    for (int r = 0; r < 8; ++r) {
      d[(size_t)r * M + col0] = acc0[r] + bv0;
      d[(size_t)r * M + col1] = acc1[r] + bv1;
    }
  }
}

// ---------------------------------------------------------------------------
// 2x2 stride-2 conv (CHW in, OIHW weights, CHW out) + exact GELU
// ---------------------------------------------------------------------------
__global__ void conv2x2s2_gelu(const float* __restrict__ x, const float* __restrict__ w,
                               const float* __restrict__ b, float* __restrict__ y,
                               int Cin, int Cout, int Hout, int Wout) {
  const int i = blockIdx.x * blockDim.x + threadIdx.x;
  const int total = Cout * Hout * Wout;
  if (i >= total) return;
  const int ox = i % Wout;
  int t = i / Wout;
  const int oy = t % Hout;
  const int oc = t / Hout;
  const int Win = Wout * 2;
  const int Hin = Hout * 2;
  float s = b[oc];
  const float* __restrict__ wp = w + (size_t)oc * Cin * 4;
  for (int ic = 0; ic < Cin; ++ic) {
    const float* __restrict__ xp = x + ((size_t)ic * Hin + oy * 2) * Win + ox * 2;
    s += wp[ic * 4 + 0] * xp[0] + wp[ic * 4 + 1] * xp[1] +
         wp[ic * 4 + 2] * xp[Win] + wp[ic * 4 + 3] * xp[Win + 1];
  }
  y[i] = 0.5f * s * (1.0f + erff(s * 0.70710678118654752440f));  // exact GELU
}

// ---------------------------------------------------------------------------
// LayerNorm over 64 channels (channels-first, in-place). One 64-thread block
// per pixel.
// ---------------------------------------------------------------------------
__global__ void ln64_chfirst(float* __restrict__ x, const float* __restrict__ w,
                             const float* __restrict__ b, int HW) {
  const int p = blockIdx.x;
  const int c = threadIdx.x;  // 0..63
  __shared__ float sm[64];
  __shared__ float s_mu, s_var;
  const size_t idx = (size_t)c * HW + p;
  const float v = x[idx];
  sm[c] = v;
  __syncthreads();
  if (c < 32) sm[c] += sm[c + 32];
  __syncthreads();
  if (c == 0) {
    float s = 0.f;
    for (int i = 0; i < 32; ++i) s += sm[i];
    s_mu = s * (1.f / 64.f);
  }
  __syncthreads();
  const float d = v - s_mu;
  __syncthreads();
  sm[c] = d * d;
  __syncthreads();
  if (c < 32) sm[c] += sm[c + 32];
  __syncthreads();
  if (c == 0) {
    float s = 0.f;
    for (int i = 0; i < 32; ++i) s += sm[i];
    s_var = s * (1.f / 64.f);
  }
  __syncthreads();
  x[idx] = w[c] * d * rsqrtf(s_var + 1e-6f) + b[c];
}

// ---------------------------------------------------------------------------
// Correlation volume: out[(y*W+x)*D + d] = x>=d ? sum_c fq[c,y,x]*fk[c,y,x-d] : 0
// Output channels-last (pixels x D) so it feeds the qkv GEMM directly.
// ---------------------------------------------------------------------------
__global__ void corr_volume(const float* __restrict__ fq, const float* __restrict__ fk,
                            float* __restrict__ out, int C, int Hc, int Wc, int D) {
  const int i = blockIdx.x * blockDim.x + threadIdx.x;
  const int total = Hc * Wc * D;
  if (i >= total) return;
  const int d = i % D;
  const int n = i / D;
  const int x = n % Wc;
  const int y = n / Wc;
  float s = 0.f;
  if (x >= d) {
    const int HWc = Hc * Wc;
    const float* __restrict__ fqp = fq + (size_t)y * Wc + x;
    const float* __restrict__ fkp = fk + (size_t)y * Wc + (x - d);
    for (int c = 0; c < C; ++c) s += fqp[(size_t)c * HWc] * fkp[(size_t)c * HWc];
  }
  out[(size_t)n * D + d] = s;
}

// ---------------------------------------------------------------------------
// 5x5 neighborhood attention core. Input: qkv (N, 3C) channels-last,
// rpb (9x9). Output (N, C). One wave32 per pixel: lanes 0..24 own the 25
// neighbors (logits + softmax via shuffles), then lanes go channel-parallel.
// ---------------------------------------------------------------------------
__global__ __launch_bounds__(256)
void na2d_attn(const float* __restrict__ qkv, const float* __restrict__ rpb,
               float* __restrict__ out, int Hn, int Wn, int C) {
  const int p = blockIdx.x * (blockDim.x >> 5) + (threadIdx.x >> 5);
  if (p >= Hn * Wn) return;
  const int lane = threadIdx.x & 31;
  const int y = p / Wn, x = p % Wn;
  int sh = y - 2; sh = sh < 0 ? 0 : (sh > Hn - 5 ? Hn - 5 : sh);
  int sw = x - 2; sw = sw < 0 ? 0 : (sw > Wn - 5 ? Wn - 5 : sw);
  const int rel_h = sh - y + 4;
  const int rel_w = sw - x + 4;
  const int C3 = 3 * C;
  const float* __restrict__ q = qkv + (size_t)p * C3;
  const float scale = rsqrtf((float)C);

  float logit = -INFINITY;
  if (lane < 25) {
    const int kh = lane / 5, kw = lane % 5;
    const int nb = (sh + kh) * Wn + (sw + kw);
    const float* __restrict__ kr = qkv + (size_t)nb * C3 + C;
    float s = 0.f;
    for (int c = 0; c < C; ++c) s += q[c] * kr[c];
    logit = s * scale + rpb[(rel_h + kh) * 9 + (rel_w + kw)];
  }
  // wave softmax over the 25 valid lanes
  float mx = logit;
  for (int o = 16; o > 0; o >>= 1) mx = fmaxf(mx, __shfl_xor(mx, o, 32));
  float e = (lane < 25) ? __expf(logit - mx) : 0.f;
  float sum = e;
  for (int o = 16; o > 0; o >>= 1) sum += __shfl_xor(sum, o, 32);
  const float a = e / sum;

  // weighted sum of V, lanes parallel over channels (coalesced per neighbor)
  for (int c0 = lane; c0 < C; c0 += 32) {
    float acc = 0.f;
#pragma unroll 5
    for (int j = 0; j < 25; ++j) {
      const float aj = __shfl(a, j, 32);
      const int nb = (sh + j / 5) * Wn + (sw + j % 5);
      acc += aj * qkv[(size_t)nb * C3 + 2 * C + c0];
    }
    out[(size_t)p * C + c0] = acc;
  }
}

// ---------------------------------------------------------------------------
// Bilinear 2x upsample, channels-last: (H,W,C) -> (2H,2W,C)
// ---------------------------------------------------------------------------
__global__ void up2x_cl(const float* __restrict__ in, float* __restrict__ out,
                        int Hn, int Wn, int C) {
  const int i = blockIdx.x * blockDim.x + threadIdx.x;
  const int total = 4 * Hn * Wn * C;
  if (i >= total) return;
  const int c = i % C;
  int t = i / C;
  const int ox = t % (2 * Wn);
  const int oy = t / (2 * Wn);
  const float sy = (float)oy * ((float)(Hn - 1) / (float)(2 * Hn - 1));
  int i0 = (int)sy; if (i0 > Hn - 2) i0 = Hn - 2;
  const float fh = sy - (float)i0;
  const float sx = (float)ox * ((float)(Wn - 1) / (float)(2 * Wn - 1));
  int j0 = (int)sx; if (j0 > Wn - 2) j0 = Wn - 2;
  const float fw = sx - (float)j0;
  const float* __restrict__ p00 = in + ((size_t)i0 * Wn + j0) * C + c;
  const float v = (1.f - fh) * ((1.f - fw) * p00[0] + fw * p00[C]) +
                  fh * ((1.f - fw) * p00[(size_t)Wn * C] + fw * p00[(size_t)Wn * C + C]);
  out[i] = v;
}

// ---------------------------------------------------------------------------
// Softmax over 128 disparities (channels-last rows), in place. Wave per pixel.
// ---------------------------------------------------------------------------
__global__ __launch_bounds__(256)
void softmax128(float* __restrict__ x, int Npix) {
  const int p = blockIdx.x * (blockDim.x >> 5) + (threadIdx.x >> 5);
  if (p >= Npix) return;
  const int lane = threadIdx.x & 31;
  float* __restrict__ row = x + (size_t)p * 128;
  float v[4];
  float m = -INFINITY;
#pragma unroll
  for (int r = 0; r < 4; ++r) { v[r] = row[lane + 32 * r]; m = fmaxf(m, v[r]); }
  for (int o = 16; o > 0; o >>= 1) m = fmaxf(m, __shfl_xor(m, o, 32));
  float s = 0.f;
#pragma unroll
  for (int r = 0; r < 4; ++r) { v[r] = __expf(v[r] - m); s += v[r]; }
  for (int o = 16; o > 0; o >>= 1) s += __shfl_xor(s, o, 32);
  const float inv = 1.f / s;
#pragma unroll
  for (int r = 0; r < 4; ++r) row[lane + 32 * r] = v[r] * inv;
}

// ---------------------------------------------------------------------------
// CHW -> HWC transpose (for guidance value path)
// ---------------------------------------------------------------------------
__global__ void chw2hwc(const float* __restrict__ in, float* __restrict__ out,
                        int C, int HW) {
  const int i = blockIdx.x * blockDim.x + threadIdx.x;
  if (i >= C * HW) return;
  const int c = i % C;
  const int n = i / C;
  out[i] = in[(size_t)c * HW + n];
}

// ---------------------------------------------------------------------------
// Deformable disparity-weighted sampling. vv is (HW, 32) channels-last,
// wts is (HW, 128) softmax weights, out is (HW, 32). One wave per pixel,
// lane == channel (32 lanes = 32 channels, fully coalesced value gathers).
// ---------------------------------------------------------------------------
__global__ __launch_bounds__(256)
void deform_sample(const float* __restrict__ vv, const float* __restrict__ wts,
                   float* __restrict__ out, int Hn, int Wn) {
  const int p = blockIdx.x * (blockDim.x >> 5) + (threadIdx.x >> 5);
  if (p >= Hn * Wn) return;
  const int lane = threadIdx.x & 31;
  const int y = p / Wn, x = p % Wn;
  const float gy = (float)y * ((float)Hn / (float)(Hn - 1)) - 0.5f;
  const float y0f = floorf(gy);
  const float fy = gy - y0f;
  const int y0 = (int)y0f;
  const float xscale = (float)Wn / (float)(Wn - 1);
  float acc = 0.f;
  const float* __restrict__ wrow = wts + (size_t)p * MAXD;
  for (int d = 0; d < MAXD; ++d) {
    const float wd = wrow[d];
    const float gx = (float)(x - d) * xscale - 0.5f;
    const float x0f = floorf(gx);
    const float fx = gx - x0f;
    const int x0 = (int)x0f;
#pragma unroll
    for (int dy = 0; dy < 2; ++dy) {
      const int yi = y0 + dy;
      if (yi < 0 || yi >= Hn) continue;
      const float cy = dy ? fy : (1.f - fy);
#pragma unroll
      for (int dx = 0; dx < 2; ++dx) {
        const int xi = x0 + dx;
        if (xi < 0 || xi >= Wn) continue;
        const float cx = dx ? fx : (1.f - fx);
        acc += wd * cy * cx * vv[((size_t)yi * Wn + xi) * 32 + lane];
      }
    }
  }
  out[(size_t)p * 32 + lane] = acc;
}

// ---------------------------------------------------------------------------
// Host orchestration
// ---------------------------------------------------------------------------
static inline void launch_gemm(const float* X, const float* Wt, const float* b,
                               float* Y, int N, int K, int M, int out_trans,
                               hipStream_t s) {
  const int waves = (N / 16) * (M / 32);   // 16x32 tile per wave
  const int blocks = CEILDIV(waves, 8);
  hipLaunchKernelGGL(wmma_gemm_kernel, dim3(blocks), dim3(256), 0, s,
                     X, Wt, b, Y, N, K, M, out_trans);
}

// Workspace layout (float offsets); all multiples of 16 floats (64B aligned).
static const size_t OFF_FQ    = 0;            // 73728   (64,24,48) CHW
static const size_t OFF_FK    = 73728;        // 73728
static const size_t OFF_TMP0  = 147456;       // 294912  conv1 out (q then k)
static const size_t OFF_CORR  = 442368;       // 36864   (1152,32) HWC
static const size_t OFF_QKV1  = 479232;       // 110592
static const size_t OFF_ATT1  = 589824;       // 36864
static const size_t OFF_X1    = 626688;       // 36864
static const size_t OFF_U1    = 663552;       // 147456  (48,96,32)
static const size_t OFF_C1    = 811008;       // 294912  (4608,64)
static const size_t OFF_QKV2  = 1105920;      // 884736
static const size_t OFF_ATT2  = 1990656;      // 294912
static const size_t OFF_X2    = 2285568;      // 294912
static const size_t OFF_U2    = 2580480;      // 1179648 (96,192,64)
static const size_t OFF_C2    = 3760128;      // 2359296 (18432,128)
static const size_t OFF_QKV3  = 6119424;      // 7077888
static const size_t OFF_ATT3  = 13197312;     // 2359296
static const size_t OFF_X3    = 15556608;     // 2359296 (softmax weights)
static const size_t OFF_GHWC  = 17915904;     // 589824  (18432,32)
static const size_t OFF_VV    = 18505728;     // 589824
static const size_t OFF_DS    = 19095552;     // 589824
// total: 19,685,376 floats ~= 78.8 MB

extern "C" void kernel_launch(void* const* d_in, const int* in_sizes, int n_in,
                              void* d_out, int out_size, void* d_ws, size_t ws_size,
                              hipStream_t stream) {
  const float* noisy    = (const float*)d_in[0];
  const float* guidance = (const float*)d_in[1];

  // Detect params flattening order: insertion-order (q.c1_w has 8192 elems at
  // index 2) vs jax tree_flatten sorted keys (k.c1_b, 64 elems, at index 2).
  const bool ins = (n_in > 2 && in_sizes[2] == 8192);
  auto P = [&](int i) { return (const float*)d_in[i]; };

  const float *q_c1w, *q_c1b, *q_ln1w, *q_ln1b, *q_c2w, *q_c2b, *q_ln2w, *q_ln2b;
  const float *k_c1w, *k_c1b, *k_ln1w, *k_ln1b, *k_c2w, *k_c2b, *k_ln2w, *k_ln2b;
  const float *na1_qkvw, *na1_qkvb, *na1_rpb, *na1_projw, *na1_projb;
  const float *na2_qkvw, *na2_qkvb, *na2_rpb, *na2_projw, *na2_projb;
  const float *na3_qkvw, *na3_qkvb, *na3_rpb, *na3_projw, *na3_projb;
  const float *up1_w, *up1_b, *up2_w, *up2_b;
  const float *vproj_w, *vproj_b, *oproj_w, *oproj_b;

  if (ins) {
    q_c1w = P(2);  q_c1b = P(3);  q_ln1w = P(4);  q_ln1b = P(5);
    q_c2w = P(6);  q_c2b = P(7);  q_ln2w = P(8);  q_ln2b = P(9);
    k_c1w = P(10); k_c1b = P(11); k_ln1w = P(12); k_ln1b = P(13);
    k_c2w = P(14); k_c2b = P(15); k_ln2w = P(16); k_ln2b = P(17);
    na1_qkvw = P(18); na1_qkvb = P(19); na1_rpb = P(20); na1_projw = P(21); na1_projb = P(22);
    up1_w = P(23); up1_b = P(24);
    na2_qkvw = P(25); na2_qkvb = P(26); na2_rpb = P(27); na2_projw = P(28); na2_projb = P(29);
    up2_w = P(30); up2_b = P(31);
    na3_qkvw = P(32); na3_qkvb = P(33); na3_rpb = P(34); na3_projw = P(35); na3_projb = P(36);
    vproj_w = P(37); vproj_b = P(38); oproj_w = P(39); oproj_b = P(40);
  } else {
    k_c1b = P(2);  k_c1w = P(3);  k_c2b = P(4);  k_c2w = P(5);
    k_ln1b = P(6); k_ln1w = P(7); k_ln2b = P(8); k_ln2w = P(9);
    na1_projb = P(10); na1_projw = P(11); na1_qkvb = P(12); na1_qkvw = P(13); na1_rpb = P(14);
    na2_projb = P(15); na2_projw = P(16); na2_qkvb = P(17); na2_qkvw = P(18); na2_rpb = P(19);
    na3_projb = P(20); na3_projw = P(21); na3_qkvb = P(22); na3_qkvw = P(23); na3_rpb = P(24);
    oproj_b = P(25); oproj_w = P(26);
    q_c1b = P(27); q_c1w = P(28); q_c2b = P(29); q_c2w = P(30);
    q_ln1b = P(31); q_ln1w = P(32); q_ln2b = P(33); q_ln2w = P(34);
    up1_b = P(35); up1_w = P(36);
    up2_b = P(37); up2_w = P(38);
    vproj_b = P(39); vproj_w = P(40);
  }

  float* ws = (float*)d_ws;
  (void)ws_size; (void)out_size;

  // ---- feature branches: conv2x2s2 + GELU + LN(64), twice ----
  {
    const int t1 = 64 * 48 * 96;   // conv1 out elems
    const int t2 = 64 * 24 * 48;   // conv2 out elems
    // q branch (noisy)
    hipLaunchKernelGGL(conv2x2s2_gelu, dim3(CEILDIV(t1, 256)), dim3(256), 0, stream,
                       noisy, q_c1w, q_c1b, ws + OFF_TMP0, 32, 64, 48, 96);
    hipLaunchKernelGGL(ln64_chfirst, dim3(48 * 96), dim3(64), 0, stream,
                       ws + OFF_TMP0, q_ln1w, q_ln1b, 48 * 96);
    hipLaunchKernelGGL(conv2x2s2_gelu, dim3(CEILDIV(t2, 256)), dim3(256), 0, stream,
                       ws + OFF_TMP0, q_c2w, q_c2b, ws + OFF_FQ, 64, 64, 24, 48);
    hipLaunchKernelGGL(ln64_chfirst, dim3(24 * 48), dim3(64), 0, stream,
                       ws + OFF_FQ, q_ln2w, q_ln2b, 24 * 48);
    // k branch (guidance)
    hipLaunchKernelGGL(conv2x2s2_gelu, dim3(CEILDIV(t1, 256)), dim3(256), 0, stream,
                       guidance, k_c1w, k_c1b, ws + OFF_TMP0, 32, 64, 48, 96);
    hipLaunchKernelGGL(ln64_chfirst, dim3(48 * 96), dim3(64), 0, stream,
                       ws + OFF_TMP0, k_ln1w, k_ln1b, 48 * 96);
    hipLaunchKernelGGL(conv2x2s2_gelu, dim3(CEILDIV(t2, 256)), dim3(256), 0, stream,
                       ws + OFF_TMP0, k_c2w, k_c2b, ws + OFF_FK, 64, 64, 24, 48);
    hipLaunchKernelGGL(ln64_chfirst, dim3(24 * 48), dim3(64), 0, stream,
                       ws + OFF_FK, k_ln2w, k_ln2b, 24 * 48);
  }

  // ---- correlation volume (channels-last) ----
  hipLaunchKernelGGL(corr_volume, dim3(CEILDIV(24 * 48 * 32, 256)), dim3(256), 0, stream,
                     ws + OFF_FQ, ws + OFF_FK, ws + OFF_CORR, 64, 24, 48, 32);

  // ---- NA stage 1 (24x48, C=32) ----
  launch_gemm(ws + OFF_CORR, na1_qkvw, na1_qkvb, ws + OFF_QKV1, 1152, 32, 96, 0, stream);
  hipLaunchKernelGGL(na2d_attn, dim3(CEILDIV(1152, 8)), dim3(256), 0, stream,
                     ws + OFF_QKV1, na1_rpb, ws + OFF_ATT1, 24, 48, 32);
  launch_gemm(ws + OFF_ATT1, na1_projw, na1_projb, ws + OFF_X1, 1152, 32, 32, 0, stream);

  // ---- up2x + 1x1 conv (GEMM) ----
  hipLaunchKernelGGL(up2x_cl, dim3(CEILDIV(4 * 1152 * 32, 256)), dim3(256), 0, stream,
                     ws + OFF_X1, ws + OFF_U1, 24, 48, 32);
  launch_gemm(ws + OFF_U1, up1_w, up1_b, ws + OFF_C1, 4608, 32, 64, 0, stream);

  // ---- NA stage 2 (48x96, C=64) ----
  launch_gemm(ws + OFF_C1, na2_qkvw, na2_qkvb, ws + OFF_QKV2, 4608, 64, 192, 0, stream);
  hipLaunchKernelGGL(na2d_attn, dim3(CEILDIV(4608, 8)), dim3(256), 0, stream,
                     ws + OFF_QKV2, na2_rpb, ws + OFF_ATT2, 48, 96, 64);
  launch_gemm(ws + OFF_ATT2, na2_projw, na2_projb, ws + OFF_X2, 4608, 64, 64, 0, stream);

  // ---- up2x + 1x1 conv ----
  hipLaunchKernelGGL(up2x_cl, dim3(CEILDIV(4 * 4608 * 64, 256)), dim3(256), 0, stream,
                     ws + OFF_X2, ws + OFF_U2, 48, 96, 64);
  launch_gemm(ws + OFF_U2, up2_w, up2_b, ws + OFF_C2, 18432, 64, 128, 0, stream);

  // ---- NA stage 3 (96x192, C=128) ----
  launch_gemm(ws + OFF_C2, na3_qkvw, na3_qkvb, ws + OFF_QKV3, 18432, 128, 384, 0, stream);
  hipLaunchKernelGGL(na2d_attn, dim3(CEILDIV(18432, 8)), dim3(256), 0, stream,
                     ws + OFF_QKV3, na3_rpb, ws + OFF_ATT3, 96, 192, 128);
  launch_gemm(ws + OFF_ATT3, na3_projw, na3_projb, ws + OFF_X3, 18432, 128, 128, 0, stream);

  // ---- softmax over disparities ----
  hipLaunchKernelGGL(softmax128, dim3(CEILDIV(18432, 8)), dim3(256), 0, stream,
                     ws + OFF_X3, 18432);

  // ---- value path: vproj, deformable sample, oproj (transposed -> CHW out) ----
  hipLaunchKernelGGL(chw2hwc, dim3(CEILDIV(32 * 18432, 256)), dim3(256), 0, stream,
                     guidance, ws + OFF_GHWC, 32, 18432);
  launch_gemm(ws + OFF_GHWC, vproj_w, vproj_b, ws + OFF_VV, 18432, 32, 32, 0, stream);
  hipLaunchKernelGGL(deform_sample, dim3(CEILDIV(18432, 8)), dim3(256), 0, stream,
                     ws + OFF_VV, ws + OFF_X3, ws + OFF_DS, 96, 192);
  launch_gemm(ws + OFF_DS, oproj_w, oproj_b, (float*)d_out, 18432, 32, 32,
              /*out_trans=*/1, stream);
}
